// DICE_6665789243401
// MI455X (gfx1250) — matrix operand order repaired and checked by reading, not compiled, not run on hardware
//
#include <hip/hip_runtime.h>
#include <hip/hip_bf16.h>

// ---------------------------------------------------------------------------
// GNN forward for MI455X (gfx1250, wave32, WMMA bf16 w/ f32 accumulate).
// All GEMMs ([rows,din]@[din,dout]) run through one WMMA kernel:
//   block = 256 threads = 8 waves; block tile 64 rows x 32 cols;
//   each wave computes one 16x16 D tile, K-loop of v_wmma_f32_16x16x32_bf16.
// A and W are staged in LDS as bf16; W is stored transposed ([n][k]) so the
// B fragment (lane n=l&15, K chunks (l>>4)*8+{0..7} and +16) is two 16B reads.
// A fragment (lane m=l&15, K chunks (l>>4)*8+{0..7} and +16) is two 16B reads.
// When A is already bf16 (second GEMM of each MLP), the A tile is staged with
// GLOBAL_LOAD_ASYNC_TO_LDS_B128 (ASYNCcnt path) instead of VGPR round-trips.
// ---------------------------------------------------------------------------

#define NN 32768
#define EE 524288
#define HID 128
#define GG 64

typedef __attribute__((ext_vector_type(16))) __bf16 v16bf;
typedef __attribute__((ext_vector_type(8)))  __bf16 bf16x8;
typedef __attribute__((ext_vector_type(8)))  float  v8f;

#define TILE_R 64
#define TILE_C 32
#define AS_STRIDE 136   // elements; 136*2B = 272B row stride (16B aligned)
#define WS_STRIDE 136

__global__ __launch_bounds__(256) void gemm_bn_wmma(
    const void* __restrict__ Aptr, int a_is_bf16,
    const float* __restrict__ W, const float* __restrict__ bias,
    float* __restrict__ Out,
    float* __restrict__ colSum, float* __restrict__ colSq, int do_stats,
    int rows, int din, int dout)
{
  __shared__ __align__(16) __bf16 As[TILE_R * AS_STRIDE];
  __shared__ __align__(16) __bf16 Ws[TILE_C * WS_STRIDE];
  __shared__ float sSum[TILE_C];
  __shared__ float sSq[TILE_C];

  const int tid = threadIdx.x;
  const int rowBase = blockIdx.x * TILE_R;   // row tiles on grid.x (can be 8192)
  const int colBase = blockIdx.y * TILE_C;   // col tiles on grid.y (<=4)
  const int kSteps = (din + 31) >> 5;
  const int KP = kSteps << 5;

  if (do_stats) {
    if (tid < TILE_C)                       sSum[tid] = 0.f;
    else if (tid < 2 * TILE_C)              sSq[tid - TILE_C] = 0.f;
  }

  // ---- stage A tile [64 x KP] into LDS as bf16 -----------------------------
  if (a_is_bf16 && din == 128 && rowBase + TILE_R <= rows) {
    // Async global->LDS copy: 64 rows x 256B = 1024 x 16B chunks, 4 per thread.
    const char* gbase = (const char*)Aptr + (long long)rowBase * 128 * 2;
#pragma unroll
    for (int i = 0; i < 4; ++i) {
      int c = tid + i * 256;          // chunk id 0..1023
      int r = c >> 4;                 // row (16 chunks per row)
      int j = c & 15;                 // 16B chunk within row
      unsigned lds = (unsigned)(unsigned long long)&As[r * AS_STRIDE + j * 8];
      unsigned long long ga = (unsigned long long)(gbase + r * 256 + j * 16);
      asm volatile("global_load_async_to_lds_b128 %0, %1, off"
                   :: "v"(lds), "v"(ga) : "memory");
    }
    asm volatile("s_wait_asynccnt 0" ::: "memory");
  } else if (a_is_bf16) {
    const __bf16* A = (const __bf16*)Aptr;
    for (int idx = tid; idx < TILE_R * KP; idx += 256) {
      int r = idx / KP, k = idx - r * KP;
      int gr = rowBase + r;
      __bf16 v = (__bf16)0.f;
      if (gr < rows && k < din) v = A[(long long)gr * din + k];
      As[r * AS_STRIDE + k] = v;
    }
  } else {
    const float* A = (const float*)Aptr;
    for (int idx = tid; idx < TILE_R * KP; idx += 256) {
      int r = idx / KP, k = idx - r * KP;
      int gr = rowBase + r;
      float v = 0.f;
      if (gr < rows && k < din) v = A[(long long)gr * din + k];
      As[r * AS_STRIDE + k] = (__bf16)v;
    }
  }
  // ---- stage W tile transposed: Ws[n][k] = W[k][colBase+n] -----------------
  for (int idx = tid; idx < TILE_C * KP; idx += 256) {
    int n = idx / KP, k = idx - n * KP;
    float v = 0.f;
    int gc = colBase + n;
    if (gc < dout && k < din) v = W[(long long)k * dout + gc];
    Ws[n * WS_STRIDE + k] = (__bf16)v;
  }
  __syncthreads();

  const int lane = tid & 31;
  const int wave = tid >> 5;
  const int wr = (wave & 3) * 16;   // wave's row sub-tile within block tile
  const int wc = (wave >> 2) * 16;  // wave's col sub-tile
  const int m  = lane & 15;
  const int hi = lane >> 4;

  v8f acc = {};
  for (int ks = 0; ks < kSteps; ++ks) {
    const int k0 = ks << 5;
    // A fragment: row m, K = hi*8+{0..7} and 16+hi*8+{0..7}
    const __bf16* ap = &As[(wr + m) * AS_STRIDE + k0 + hi * 8];
    bf16x8 a0 = *(const bf16x8*)ap;
    bf16x8 a1 = *(const bf16x8*)(ap + 16);
    v16bf a;
#pragma unroll
    for (int i = 0; i < 8; ++i) { a[i] = a0[i]; a[i + 8] = a1[i]; }
    // B fragment: col m, K = hi*8+{0..7} and 16+hi*8+{0..7} (Ws is [n][k])
    const __bf16* bp = &Ws[(wc + m) * WS_STRIDE + k0 + hi * 8];
    bf16x8 b0 = *(const bf16x8*)bp;
    bf16x8 b1 = *(const bf16x8*)(bp + 16);
    v16bf b;
#pragma unroll
    for (int i = 0; i < 8; ++i) { b[i] = b0[i]; b[i + 8] = b1[i]; }

    acc = __builtin_amdgcn_wmma_f32_16x16x32_bf16(
        /*neg_a=*/false, a, /*neg_b=*/false, b,
        /*c_mod=*/(short)0, acc, /*reuse_a=*/false, /*reuse_b=*/false);
  }

  // ---- epilogue: +bias, store f32, fused BN statistics ---------------------
  const int gc = colBase + wc + m;
  const float bv = (gc < dout) ? bias[gc] : 0.f;
  float psum = 0.f, psq = 0.f;
#pragma unroll
  for (int v = 0; v < 8; ++v) {
    int gr = rowBase + wr + hi * 8 + v;
    float h = acc[v] + bv;
    if (gr < rows && gc < dout) {
      Out[(long long)gr * dout + gc] = h;
      psum += h; psq += h * h;
    }
  }
  if (do_stats) {
    atomicAdd(&sSum[wc + m], psum);   // ds_add_f32 (column-local)
    atomicAdd(&sSq[wc + m],  psq);
    __syncthreads();
    if (tid < TILE_C) {
      int c = colBase + tid;
      if (c < dout) {
        atomicAdd(&colSum[c], sSum[tid]);
        atomicAdd(&colSq[c],  sSq[tid]);
      }
    }
  }
}

__global__ void bn_finalize(const float* __restrict__ colSum,
                            const float* __restrict__ colSq,
                            float* __restrict__ mean, float* __restrict__ rstd,
                            int rows, int dh)
{
  int c = blockIdx.x * blockDim.x + threadIdx.x;
  if (c < dh) {
    float inv = 1.f / (float)rows;
    float mu  = colSum[c] * inv;
    float var = colSq[c] * inv - mu * mu;
    mean[c] = mu;
    rstd[c] = rsqrtf(var + 1e-5f);
  }
}

// BN (batch stats) + tanh-approx GELU; writes bf16 for the second GEMM's A.
__global__ void bn_gelu_bf16(const float* __restrict__ H,
                             const float* __restrict__ mean,
                             const float* __restrict__ rstd,
                             const float* __restrict__ g,
                             const float* __restrict__ bt,
                             __bf16* __restrict__ Hb, long long total)
{
  long long i = blockIdx.x * (long long)blockDim.x + threadIdx.x;
  if (i < total) {
    int c = (int)(i & (HID - 1));
    float v = (H[i] - mean[c]) * rstd[c] * g[c] + bt[c];
    float t = tanhf(0.7978845608028654f * (v + 0.044715f * v * v * v));
    Hb[i] = (__bf16)(0.5f * v * (1.f + t));
  }
}

// msg = nh[src] * eh ; nz[dst] += msg   (segment_sum via f32 atomics; nz fits L2)
__global__ void edge_scatter(const float* __restrict__ nh,
                             const float* __restrict__ eh,
                             const int* __restrict__ src,
                             const int* __restrict__ dst,
                             float* __restrict__ nz, long long total)
{
  long long i = blockIdx.x * (long long)blockDim.x + threadIdx.x;
  if (i < total) {
    int e = (int)(i >> 7), h = (int)(i & (HID - 1));
    float m = nh[(long long)src[e] * HID + h] * eh[i];
    atomicAdd(&nz[(long long)dst[e] * HID + h], m);
  }
}

__global__ void node_update(const float* __restrict__ nh,
                            const float* __restrict__ nz,
                            const float* __restrict__ epsP,
                            float* __restrict__ out, long long total)
{
  long long i = blockIdx.x * (long long)blockDim.x + threadIdx.x;
  if (i < total) out[i] = (1.f + epsP[0]) * nh[i] + nz[i];
}

__global__ void edge_update(const float* __restrict__ eh,
                            const float* __restrict__ nz,
                            const int* __restrict__ src,
                            const int* __restrict__ dst,
                            const float* __restrict__ epsP,
                            float* __restrict__ out, long long total)
{
  long long i = blockIdx.x * (long long)blockDim.x + threadIdx.x;
  if (i < total) {
    int e = (int)(i >> 7), h = (int)(i & (HID - 1));
    out[i] = (1.f + epsP[0]) * eh[i]
           + nz[(long long)src[e] * HID + h]
           - nz[(long long)dst[e] * HID + h];
  }
}

__global__ void pool_nodes(const float* __restrict__ nh,
                           const int* __restrict__ batch,
                           float* __restrict__ gh, long long total)
{
  long long i = blockIdx.x * (long long)blockDim.x + threadIdx.x;
  if (i < total) {
    int n = (int)(i >> 7), h = (int)(i & (HID - 1));
    atomicAdd(&gh[(long long)batch[n] * HID + h], nh[i]);
  }
}

__global__ void pool_edges(const float* __restrict__ eh,
                           const int* __restrict__ src,
                           const int* __restrict__ batch,
                           float* __restrict__ gh, long long total)
{
  long long i = blockIdx.x * (long long)blockDim.x + threadIdx.x;
  if (i < total) {
    int e = (int)(i >> 7), h = (int)(i & (HID - 1));
    atomicAdd(&gh[(long long)batch[src[e]] * HID + h], eh[i]);
  }
}

// one_hot(y, C)[:,:,None] * feat[:,None,:]  flattened -> [rows, C*F]
__global__ void onehot_outer(const float* __restrict__ feat,
                             const int* __restrict__ y,
                             float* __restrict__ out,
                             int C, int F, long long total)
{
  long long i = blockIdx.x * (long long)blockDim.x + threadIdx.x;
  if (i < total) {
    int cf = C * F;
    long long r = i / cf;
    int rem = (int)(i - r * cf);
    int cls = rem / F, j = rem - cls * F;
    out[i] = (cls == y[r]) ? feat[r * F + j] : 0.f;
  }
}

// ---------------------------------------------------------------------------
// Host-side orchestration
// ---------------------------------------------------------------------------
struct MLPW { const float *W1, *W2, *b1, *b2, *bt, *g; int din, dout; };

static inline MLPW mk_mlp(void* const* d, int base, int din, int dout) {
  // leaves within one MLP dict in sorted-key order: W1, W2, b1, b2, bt, g
  MLPW p;
  p.W1 = (const float*)d[base + 0];
  p.W2 = (const float*)d[base + 1];
  p.b1 = (const float*)d[base + 2];
  p.b2 = (const float*)d[base + 3];
  p.bt = (const float*)d[base + 4];
  p.g  = (const float*)d[base + 5];
  p.din = din; p.dout = dout;
  return p;
}

static void run_mlp(const void* A, int a_bf16, long long rows, const MLPW& p,
                    float* OUT, float* H, __bf16* Hb, float* stats,
                    hipStream_t stream)
{
  float* colSum = stats;        float* colSq = stats + HID;
  float* mean   = stats + 256;  float* rstd  = stats + 384;
  hipMemsetAsync(colSum, 0, 2 * HID * sizeof(float), stream);

  dim3 blk(256);
  dim3 g1((unsigned)((rows + TILE_R - 1) / TILE_R), (HID + TILE_C - 1) / TILE_C);
  gemm_bn_wmma<<<g1, blk, 0, stream>>>(A, a_bf16, p.W1, p.b1, H,
                                       colSum, colSq, 1, (int)rows, p.din, HID);
  bn_finalize<<<1, HID, 0, stream>>>(colSum, colSq, mean, rstd, (int)rows, HID);

  long long total = rows * HID;
  bn_gelu_bf16<<<(unsigned)((total + 255) / 256), 256, 0, stream>>>(
      H, mean, rstd, p.g, p.bt, Hb, total);

  dim3 g2((unsigned)((rows + TILE_R - 1) / TILE_R), (p.dout + TILE_C - 1) / TILE_C);
  gemm_bn_wmma<<<g2, blk, 0, stream>>>(Hb, 1, p.W2, p.b2, OUT,
                                       nullptr, nullptr, 0, (int)rows, HID, p.dout);
}

extern "C" void kernel_launch(void* const* d_in, const int* in_sizes, int n_in,
                              void* d_out, int out_size, void* d_ws, size_t ws_size,
                              hipStream_t stream)
{
  (void)in_sizes; (void)n_in; (void)out_size; (void)ws_size;

  // JAX pytree flatten order (sorted dict keys, recursively):
  // 0:batch 1:edge_attr 2:edge_index 3:edge_y 4:node_y 5..62:params 63:x
  const int*   batch      = (const int*)  d_in[0];
  const float* edge_attr  = (const float*)d_in[1];
  const int*   edge_index = (const int*)  d_in[2];
  const int*   edge_y     = (const int*)  d_in[3];
  const int*   node_y     = (const int*)  d_in[4];
  const float* x          = (const float*)d_in[63];

  // params leaves (sorted): gnn[0]{ef(6),ef_eps,nf(6),nf_eps} gnn[1]{...}
  //                         init_ef(6) init_nf(6) out_ef(6) out_gf(6) out_nf(6)
  MLPW gnn_ef[2] = { mk_mlp(d_in, 5,  HID, HID), mk_mlp(d_in, 19, HID, HID) };
  const float* ef_eps[2] = { (const float*)d_in[11], (const float*)d_in[25] };
  MLPW gnn_nf[2] = { mk_mlp(d_in, 12, HID, HID), mk_mlp(d_in, 26, HID, HID) };
  const float* nf_eps[2] = { (const float*)d_in[18], (const float*)d_in[32] };
  MLPW init_ef = mk_mlp(d_in, 33, 8,   HID);
  MLPW init_nf = mk_mlp(d_in, 39, 16,  HID);
  MLPW out_ef  = mk_mlp(d_in, 45, HID, 16);
  MLPW out_gf  = mk_mlp(d_in, 51, HID, 32);
  MLPW out_nf  = mk_mlp(d_in, 57, HID, 16);

  const int* src = edge_index;        // row 0
  const int* dst = edge_index + EE;   // row 1

  // ---- workspace partition -------------------------------------------------
  char* cur = (char*)d_ws;
  auto wsalloc = [&](size_t bytes) -> void* {
    void* p = cur; cur += (bytes + 255) & ~(size_t)255; return p;
  };
  float*  nh    = (float*)wsalloc((size_t)NN * HID * 4);
  float*  tmpN  = (float*)wsalloc((size_t)NN * HID * 4);
  float*  nz    = (float*)wsalloc((size_t)NN * HID * 4);
  float*  eh    = (float*)wsalloc((size_t)EE * HID * 4);
  float*  tmpE  = (float*)wsalloc((size_t)EE * HID * 4);
  float*  H     = (float*)wsalloc((size_t)EE * HID * 4);         // MLP hidden
  __bf16* Hb    = (__bf16*)wsalloc((size_t)EE * HID * 2);        // GELU out (bf16)
  float*  stats = (float*)wsalloc(512 * 4);
  float*  ghbuf = (float*)wsalloc((size_t)GG * HID * 4);
  float*  nho   = (float*)wsalloc((size_t)NN * 16 * 4);
  float*  eho   = (float*)wsalloc((size_t)EE * 16 * 4);

  // ---- encoders ------------------------------------------------------------
  run_mlp(x,         0, NN, init_nf, nh, H, Hb, stats, stream);
  run_mlp(edge_attr, 0, EE, init_ef, eh, H, Hb, stats, stream);

  // ---- GIN layers ----------------------------------------------------------
  const long long totN = (long long)NN * HID, totE = (long long)EE * HID;
  for (int l = 0; l < 2; ++l) {
    hipMemsetAsync(nz, 0, (size_t)NN * HID * 4, stream);
    edge_scatter<<<(unsigned)((totE + 255) / 256), 256, 0, stream>>>(
        nh, eh, src, dst, nz, totE);
    node_update<<<(unsigned)((totN + 255) / 256), 256, 0, stream>>>(
        nh, nz, nf_eps[l], tmpN, totN);
    edge_update<<<(unsigned)((totE + 255) / 256), 256, 0, stream>>>(
        eh, nz, src, dst, ef_eps[l], tmpE, totE);
    run_mlp(tmpN, 0, NN, gnn_nf[l], nh, H, Hb, stats, stream);
    run_mlp(tmpE, 0, EE, gnn_ef[l], eh, H, Hb, stats, stream);
  }

  // ---- graph head ----------------------------------------------------------
  float* out_nh_region = (float*)d_out;
  float* out_eh_region = (float*)d_out + (long long)NN * 256;
  float* out_gh_region = (float*)d_out + (long long)NN * 256 + (long long)EE * 128;

  hipMemsetAsync(ghbuf, 0, (size_t)GG * HID * 4, stream);
  pool_nodes<<<(unsigned)((totN + 255) / 256), 256, 0, stream>>>(nh, batch, ghbuf, totN);
  pool_edges<<<(unsigned)((totE + 255) / 256), 256, 0, stream>>>(eh, src, batch, ghbuf, totE);
  run_mlp(ghbuf, 0, GG, out_gf, out_gh_region, H, Hb, stats, stream);

  // ---- node / edge heads ---------------------------------------------------
  run_mlp(nh, 0, NN, out_nf, nho, H, Hb, stats, stream);
  run_mlp(eh, 0, EE, out_ef, eho, H, Hb, stats, stream);

  long long totNO = (long long)NN * 16 * 16;   // [N, NF_IN*NF_OUT]
  long long totEO = (long long)EE * 8 * 16;    // [E, EF_IN*EF_OUT]
  onehot_outer<<<(unsigned)((totNO + 255) / 256), 256, 0, stream>>>(
      nho, node_y, out_nh_region, 16, 16, totNO);
  onehot_outer<<<(unsigned)((totEO + 255) / 256), 256, 0, stream>>>(
      eho, edge_y, out_eh_region, 8, 16, totEO);
}
